// GNN_19069654794768
// MI455X (gfx1250) — matrix operand order repaired and checked by reading, not compiled, not run on hardware
//
#include <hip/hip_runtime.h>

#define NNODES  100000
#define NEDGES  1600000
#define NGRAPH  512
#define HSTRIDE 64          // padded feature stride for h buffers

typedef __attribute__((ext_vector_type(16))) _Float16 v16h;
typedef __attribute__((ext_vector_type(8)))  float    v8f;

// ---------------------------------------------------------------------------
// WMMA fragment layouts (wave32), CDNA5 ISA §7.12.2:
//  A 16x32 f16 : lane holds row m=lane&15; element e -> K = e + 16*(lane>>4) (this file's order, not ISA's)
//  B 32x16 f16 : lane holds col n=lane&15; element e -> K = e + 16*(lane>>4) (same as A; not ISA's)
//  C/D 16x16 f32: lane holds col n=lane&15; vgpr v -> M = v + 8*(lane>>4)
// B fragments are PRE-SWIZZLED in workspace: frag is 512 f16 = 32 lanes x 16
// contiguous halves, so a lane loads its register image as one 32B vector.
// ---------------------------------------------------------------------------
__device__ inline v16h load_a_frag(const _Float16* tile, int ldk, int kbase, int lane) {
  int m  = lane & 15;
  int h8 = (lane >> 4) * 8;
  v16h a;
#pragma unroll
  for (int e = 0; e < 16; ++e) {
    int k = kbase + e + 2 * h8;
    a[e] = tile[m * ldk + k];
  }
  return a;
}

__device__ inline v16h load_b_sw(const _Float16* frag, int lane) {
  return *(const v16h*)(frag + lane * 16);     // 32B contiguous per lane
}

#define WMMA_F16(a, b, c) \
  __builtin_amdgcn_wmma_f32_16x16x32_f16(false, (a), false, (b), (short)0, (c), false, false)

// Per-MLP packed weight block in workspace (8192 bytes each):
//   [0,4096)    W1 swizzled frags: fi = kc*2+nh, fi*512 f16 each (BN-folded, K-padded)
//   [4096,4224) b1eff f32 [32] (BN folded)
//   [4224,6272) W2 swizzled frags: nh*512 f16 each
//   [6272,6400) b2 f32 [32]
#define WBLK_BYTES 8192

struct MlpSrc { const float *W1, *b1, *g, *bt, *W2, *b2; int K; };
struct PrepArgs { MlpSrc m[6]; };

__global__ void prep_kernel(PrepArgs args, char* wbase) {
  int b = blockIdx.x;
  MlpSrc s = args.m[b];
  char* blk = wbase + (size_t)b * WBLK_BYTES;
  _Float16* W1d = (_Float16*)blk;
  float*    b1d = (float*)(blk + 4096);
  _Float16* W2d = (_Float16*)(blk + 4224);
  float*    b2d = (float*)(blk + 6272);
  const float inv = rsqrtf(1.0f + 1e-5f);   // BN eval: running_mean=0, var=1

  // W1 -> 4 swizzled fragments (kc in {0,1}, nh in {0,1})
  for (int i = threadIdx.x; i < 2048; i += blockDim.x) {
    int fi = i >> 9, p = i & 511;
    int lane = p >> 4, e = p & 15;
    int kc = fi >> 1, nh = fi & 1;
    int k = kc * 32 + e + 16 * (lane >> 4);
    int n = nh * 16 + (lane & 15);
    float v = (k < s.K) ? s.W1[k * 32 + n] * (s.g[n] * inv) : 0.0f;
    W1d[i] = (_Float16)v;
  }
  // W2 -> 2 swizzled fragments (nh in {0,1})
  for (int i = threadIdx.x; i < 1024; i += blockDim.x) {
    int nh = i >> 9, p = i & 511;
    int lane = p >> 4, e = p & 15;
    int k = e + 16 * (lane >> 4);
    int n = nh * 16 + (lane & 15);
    W2d[i] = (_Float16)s.W2[k * 32 + n];
  }
  for (int j = threadIdx.x; j < 32; j += blockDim.x) {
    b1d[j] = s.b1[j] * (s.g[j] * inv) + s.bt[j];
    b2d[j] = s.b2[j];
  }
}

// ---------------------------------------------------------------------------
// Shared tail: Lin1 C-frags -> bias+ReLU -> LDS -> Lin2 WMMA -> bias(+ReLU)
// ---------------------------------------------------------------------------
__device__ inline void mlp_tail(v8f c0, v8f c1, _Float16* ht,
                                const char* wblk, int lane, int node0,
                                float* __restrict__ hout, bool relu_out) {
  const float*    b1 = (const float*)(wblk + 4096);
  const _Float16* W2 = (const _Float16*)(wblk + 4224);
  const float*    b2 = (const float*)(wblk + 6272);

  const int n  = lane & 15;
  const int mh = (lane >> 4) * 8;
  float bb0 = b1[n], bb1 = b1[n + 16];
#pragma unroll
  for (int v = 0; v < 8; ++v) {
    float h0 = c0[v] + bb0; h0 = h0 > 0.f ? h0 : 0.f;
    float h1 = c1[v] + bb1; h1 = h1 > 0.f ? h1 : 0.f;
    ht[(mh + v) * 32 + n]      = (_Float16)h0;
    ht[(mh + v) * 32 + n + 16] = (_Float16)h1;
  }
  __syncthreads();

  v16h ah  = load_a_frag(ht, 32, 0, lane);
  v16h b20 = load_b_sw(W2, lane);
  v16h b21 = load_b_sw(W2 + 512, lane);
  v8f d0 = {}, d1 = {};
  d0 = WMMA_F16(ah, b20, d0);
  d1 = WMMA_F16(ah, b21, d1);

  float cb0 = b2[n], cb1 = b2[n + 16];
#pragma unroll
  for (int v = 0; v < 8; ++v) {
    int row = node0 + mh + v;
    float o0 = d0[v] + cb0;
    float o1 = d1[v] + cb1;
    if (relu_out) { o0 = o0 > 0.f ? o0 : 0.f; o1 = o1 > 0.f ? o1 : 0.f; }
    hout[row * HSTRIDE + n]      = o0;
    hout[row * HSTRIDE + n + 16] = o1;
  }
}

// ---------------------------------------------------------------------------
// Encoder: h0[n][0:32] = mlp_enc(x[n]), [32:52]=pe, [52:64]=0. 1 wave per 16 nodes.
// ---------------------------------------------------------------------------
__global__ __launch_bounds__(32) void enc_kernel(const float* __restrict__ x,
                                                 const float* __restrict__ pe,
                                                 const char* __restrict__ wblk,
                                                 float* __restrict__ hout) {
  __shared__ __align__(16) _Float16 zt[16 * 64];
  __shared__ __align__(16) _Float16 ht[16 * 32];
  const int lane  = threadIdx.x;
  const int node0 = blockIdx.x * 16;

  // stage x tile (K 37 -> pad 64) into LDS as f16 (x stride 37: scalar path)
#pragma unroll
  for (int i = 0; i < 32; ++i) {
    int idx = lane + 32 * i;
    int r = idx >> 6, c = idx & 63;
    float v = (c < 37) ? x[(node0 + r) * 37 + c] : 0.0f;
    zt[idx] = (_Float16)v;
  }
  __syncthreads();

  const _Float16* W1 = (const _Float16*)wblk;
  v8f c0 = {}, c1 = {};
#pragma unroll
  for (int kc = 0; kc < 2; ++kc) {
    v16h a  = load_a_frag(zt, 64, kc * 32, lane);
    v16h bl = load_b_sw(W1 + (kc * 2 + 0) * 512, lane);
    v16h br = load_b_sw(W1 + (kc * 2 + 1) * 512, lane);
    c0 = WMMA_F16(a, bl, c0);
    c1 = WMMA_F16(a, br, c1);
  }
  mlp_tail(c0, c1, ht, wblk, lane, node0, hout, /*relu_out=*/false);

  // append PE (cols 32..51) and zero pad (52..63)
#pragma unroll
  for (int i = 0; i < 16; ++i) {
    int idx = lane + 32 * i;
    int r = idx >> 5, c = idx & 31;
    float v = (c < 20) ? pe[(node0 + r) * 20 + c] : 0.0f;
    hout[(node0 + r) * HSTRIDE + 32 + c] = v;
  }
}

// ---------------------------------------------------------------------------
// GIN layer: hout = relu( MLP( hin + agg ) ). K1 = 64 (layer 0) or 32.
// ---------------------------------------------------------------------------
template <int K1>
__global__ __launch_bounds__(32) void gin_kernel(const float* __restrict__ hin,
                                                 const float* __restrict__ agg,
                                                 const char* __restrict__ wblk,
                                                 float* __restrict__ hout) {
  __shared__ __align__(16) _Float16 zt[16 * 64];
  __shared__ __align__(16) _Float16 ht[16 * 32];
  const int lane  = threadIdx.x;
  const int node0 = blockIdx.x * 16;

  // stage z = hin + agg as f16 into LDS: float4 reads, 8B packed LDS stores
  constexpr int C4   = K1 / 4;          // float4 groups per row
  constexpr int NITR = 16 * C4 / 32;    // groups per lane
#pragma unroll
  for (int i = 0; i < NITR; ++i) {
    int q  = lane + 32 * i;
    int r  = q / C4;
    int c4 = q - r * C4;
    const float4* hp = (const float4*)(hin + (node0 + r) * HSTRIDE) + c4;
    const float4* ap = (const float4*)(agg + (node0 + r) * HSTRIDE) + c4;
    float4 hv = *hp, av = *ap;
    union { _Float16 h[4]; uint2 u; } pk;
    pk.h[0] = (_Float16)(hv.x + av.x);
    pk.h[1] = (_Float16)(hv.y + av.y);
    pk.h[2] = (_Float16)(hv.z + av.z);
    pk.h[3] = (_Float16)(hv.w + av.w);
    *(uint2*)&zt[r * 64 + c4 * 4] = pk.u;
  }
  __syncthreads();

  const _Float16* W1 = (const _Float16*)wblk;
  v8f c0 = {}, c1 = {};
#pragma unroll
  for (int kc = 0; kc < K1 / 32; ++kc) {
    v16h a  = load_a_frag(zt, 64, kc * 32, lane);
    v16h bl = load_b_sw(W1 + (kc * 2 + 0) * 512, lane);
    v16h br = load_b_sw(W1 + (kc * 2 + 1) * 512, lane);
    c0 = WMMA_F16(a, bl, c0);
    c1 = WMMA_F16(a, br, c1);
  }
  mlp_tail(c0, c1, ht, wblk, lane, node0, hout, /*relu_out=*/true);
}

// ---------------------------------------------------------------------------
// Edge aggregation: agg[dst] += h[src]. One float4 read + 4 FP32 atomics per
// thread; edge-index loads amortized 4x, all traffic L2-resident.
// ---------------------------------------------------------------------------
template <int F>
__global__ void scatter_kernel(const float* __restrict__ h, float* __restrict__ agg,
                               const int* __restrict__ src, const int* __restrict__ dst) {
  constexpr int C = F / 4;               // float4 chunks per edge (13 or 8)
  const int total = NEDGES * C;
  for (int idx = blockIdx.x * blockDim.x + threadIdx.x; idx < total;
       idx += gridDim.x * blockDim.x) {
    int e = idx / C;
    int c = idx - e * C;
    const float4 v = *((const float4*)(h + src[e] * HSTRIDE) + c);
    float* a = agg + dst[e] * HSTRIDE + c * 4;
    unsafeAtomicAdd(a + 0, v.x);
    unsafeAtomicAdd(a + 1, v.y);
    unsafeAtomicAdd(a + 2, v.z);
    unsafeAtomicAdd(a + 3, v.w);
  }
}

__global__ void zero_kernel(float* __restrict__ p, int n) {
  for (int i = blockIdx.x * blockDim.x + threadIdx.x; i < n; i += gridDim.x * blockDim.x)
    p[i] = 0.0f;
}

__global__ void pool_kernel(const float* __restrict__ h, const int* __restrict__ bid,
                            float* __restrict__ pooled) {
  const int total = NNODES * 8;          // 8 float4 chunks per node (32 feats)
  for (int idx = blockIdx.x * blockDim.x + threadIdx.x; idx < total;
       idx += gridDim.x * blockDim.x) {
    int nd = idx >> 3;
    int c  = idx & 7;
    const float4 v = *((const float4*)(h + nd * HSTRIDE) + c);
    float* p = pooled + bid[nd] * 32 + c * 4;
    unsafeAtomicAdd(p + 0, v.x);
    unsafeAtomicAdd(p + 1, v.y);
    unsafeAtomicAdd(p + 2, v.z);
    unsafeAtomicAdd(p + 3, v.w);
  }
}

__global__ void head_kernel(const float* __restrict__ pooled,
                            const float* __restrict__ W1, const float* __restrict__ b1,
                            const float* __restrict__ W2, const float* __restrict__ b2,
                            float* __restrict__ out) {
  int g = blockIdx.x * blockDim.x + threadIdx.x;
  if (g >= NGRAPH) return;
  float hid[32];
#pragma unroll 4
  for (int j = 0; j < 32; ++j) {
    float s = b1[j];
    for (int k = 0; k < 32; ++k) s += pooled[g * 32 + k] * W1[k * 32 + j];
    hid[j] = s > 0.f ? s : 0.f;
  }
#pragma unroll
  for (int o = 0; o < 2; ++o) {
    float s = b2[o];
    for (int j = 0; j < 32; ++j) s += hid[j] * W2[j * 2 + o];
    out[g * 2 + o] = s;
  }
}

// ---------------------------------------------------------------------------
extern "C" void kernel_launch(void* const* d_in, const int* in_sizes, int n_in,
                              void* d_out, int out_size, void* d_ws, size_t ws_size,
                              hipStream_t stream) {
  const float* x   = (const float*)d_in[0];
  const float* pe  = (const float*)d_in[1];
  const int*   ei  = (const int*)d_in[2];
  const int*   bid = (const int*)d_in[3];
  const int* src = ei;
  const int* dst = ei + NEDGES;

  // Param pytree leaves start at index 4. Detect flatten order deterministically
  // from in_sizes: second leaf of an MLP block is a matrix (>=64) for
  // sorted-key order {W1,W2,b1,b2,bt,g}, or a 32-vector for insertion order
  // {W1,b1,g,bt,W2,b2}.
  auto mlp = [&](int base, int K) -> MlpSrc {
    MlpSrc s; s.K = K;
    if (in_sizes[base + 1] >= 64) {                 // sorted keys
      s.W1 = (const float*)d_in[base + 0];
      s.W2 = (const float*)d_in[base + 1];
      s.b1 = (const float*)d_in[base + 2];
      s.b2 = (const float*)d_in[base + 3];
      s.bt = (const float*)d_in[base + 4];
      s.g  = (const float*)d_in[base + 5];
    } else {                                        // insertion order
      s.W1 = (const float*)d_in[base + 0];
      s.b1 = (const float*)d_in[base + 1];
      s.g  = (const float*)d_in[base + 2];
      s.bt = (const float*)d_in[base + 3];
      s.W2 = (const float*)d_in[base + 4];
      s.b2 = (const float*)d_in[base + 5];
    }
    return s;
  };

  PrepArgs pa;
  pa.m[0] = mlp(4, 37);                 // encoder
  pa.m[1] = mlp(10, 52);                // gin layer 0
  for (int l = 1; l < 5; ++l) pa.m[1 + l] = mlp(10 + 6 * l, 32);

  const float *hW1, *hb1, *hW2, *hb2;   // head: {W1,W2,b1,b2} vs {W1,b1,W2,b2}
  if (in_sizes[41] == 64) {
    hW1 = (const float*)d_in[40]; hW2 = (const float*)d_in[41];
    hb1 = (const float*)d_in[42]; hb2 = (const float*)d_in[43];
  } else {
    hW1 = (const float*)d_in[40]; hb1 = (const float*)d_in[41];
    hW2 = (const float*)d_in[42]; hb2 = (const float*)d_in[43];
  }

  // workspace layout
  char*  ws     = (char*)d_ws;
  char*  wblk   = ws;                                   // 6 * 8192 B
  float* h0     = (float*)(ws + 65536);
  float* h1     = h0 + (size_t)NNODES * HSTRIDE;
  float* agg    = h1 + (size_t)NNODES * HSTRIDE;
  float* pooled = agg + (size_t)NNODES * HSTRIDE;

  prep_kernel<<<6, 256, 0, stream>>>(pa, wblk);
  enc_kernel<<<NNODES / 16, 32, 0, stream>>>(x, pe, wblk, h0);

  float* hin = h0;
  float* hout = h1;
  for (int l = 0; l < 5; ++l) {
    zero_kernel<<<2048, 256, 0, stream>>>(agg, NNODES * HSTRIDE);
    if (l == 0) {
      scatter_kernel<52><<<8192, 256, 0, stream>>>(hin, agg, src, dst);
      gin_kernel<64><<<NNODES / 16, 32, 0, stream>>>(hin, agg,
          wblk + (size_t)1 * WBLK_BYTES, hout);
    } else {
      scatter_kernel<32><<<8192, 256, 0, stream>>>(hin, agg, src, dst);
      gin_kernel<32><<<NNODES / 16, 32, 0, stream>>>(hin, agg,
          wblk + (size_t)(l + 1) * WBLK_BYTES, hout);
    }
    float* t = hin; hin = hout; hout = t;
  }

  zero_kernel<<<64, 256, 0, stream>>>(pooled, NGRAPH * 32);
  pool_kernel<<<4096, 256, 0, stream>>>(hin, bid, pooled);
  head_kernel<<<(NGRAPH + 255) / 256, 256, 0, stream>>>(pooled, hW1, hb1, hW2, hb2,
                                                        (float*)d_out);
}
